// GSER_71571335021188
// MI455X (gfx1250) — matrix operand order repaired and checked by reading, not compiled, not run on hardware
//
#include <hip/hip_runtime.h>

// ---------------------------------------------------------------------------
// GSER fused reservoir update for MI455X (gfx1250, wave32, WMMA bf16).
// B=4096, IN=512, A=2048, MAXD=4096.  ~68.8 GFLOP of GEMM fused with the
// gate/tanh/threshold epilogue; bf16 WMMA (16x16x32) with f32 accumulation.
// Double-buffered LDS, CDNA5 async global->LDS copies (ASYNCcnt), two tight
// pipelined loops with a peeled phase boundary.
// ---------------------------------------------------------------------------

typedef __bf16 bf16_t;
typedef __attribute__((ext_vector_type(16))) __bf16 v16bf;
typedef __attribute__((ext_vector_type(8)))  float  v8f;
typedef __attribute__((ext_vector_type(4)))  int    int4v;

#define BDIM  4096   // batch
#define INDIM 512    // input dim
#define AD    2048   // active dim
#define MAXD  4096   // padded state dim

#define BM 128       // workgroup M tile (8 waves x 16 rows)
#define BN 64        // workgroup N tile (4 WMMA tiles wide)
#define KT 32        // K step (bf16 WMMA K)
#define LDSROW 40    // padded LDS row pitch in bf16 (80B: 16B-aligned, bank-skewed)
#define NT1 (INDIM / KT)          // 16 phase-1 K-steps
#define NT2 (AD / KT)             // 64 phase-2 K-steps

#define WMMA_BF16(a, b, c) \
  __builtin_amdgcn_wmma_f32_16x16x32_bf16(false, (a), false, (b), (short)0, (c), false, false)

#if defined(__has_builtin)
#if __has_builtin(__builtin_amdgcn_global_load_async_to_lds_b128) && \
    __has_builtin(__builtin_amdgcn_s_wait_asynccnt)
#define GSER_ASYNC_COPY 1
#endif
#endif
#ifndef GSER_ASYNC_COPY
#define GSER_ASYNC_COPY 0
#endif

#if GSER_ASYNC_COPY
// Builtin signature (confirmed by toolchain diagnostic):
//   void (addrspace(1) int4*, addrspace(3) int4*, imm int offset, imm int cpol)
// flat->LDS addrspacecast is low-32-bit truncation on AMDGPU; flat->global is
// identity bits. Go through uintptr_t so the casts always compile.
#define AS_GLOBAL(p) ((__attribute__((address_space(1))) int4v*)(unsigned long long)(uintptr_t)(p))
#define AS_LOCAL(p)  ((__attribute__((address_space(3))) int4v*)(unsigned int)(uintptr_t)(p))
#endif

__device__ __forceinline__ float sigmoidf_(float x) { return 1.0f / (1.0f + __expf(-x)); }
__device__ __forceinline__ float softplusf_(float x) { return log1pf(__expf(x)); }

// ---- 32B / 64B global->LDS copies, split into issue (before compute) and
// ---- commit (after compute) so latency overlaps the WMMA work.
__device__ __forceinline__ void cp32_issue(bf16_t* dst, const bf16_t* src, uint4* st) {
#if GSER_ASYNC_COPY
  __builtin_amdgcn_global_load_async_to_lds_b128(AS_GLOBAL(src), AS_LOCAL(dst), 0, 0);
  __builtin_amdgcn_global_load_async_to_lds_b128(AS_GLOBAL(src + 8), AS_LOCAL(dst + 8), 0, 0);
  (void)st;
#else
  const uint4* g = (const uint4*)src;
  st[0] = g[0]; st[1] = g[1];
  (void)dst;
#endif
}
__device__ __forceinline__ void cp32_commit(bf16_t* dst, const uint4* st) {
#if !GSER_ASYNC_COPY
  uint4* l = (uint4*)dst;
  l[0] = st[0]; l[1] = st[1];
#else
  (void)dst; (void)st;
#endif
}
__device__ __forceinline__ void cp64_issue(bf16_t* dst, const bf16_t* src, uint4* st) {
#if GSER_ASYNC_COPY
  __builtin_amdgcn_global_load_async_to_lds_b128(AS_GLOBAL(src),      AS_LOCAL(dst), 0, 0);
  __builtin_amdgcn_global_load_async_to_lds_b128(AS_GLOBAL(src + 8),  AS_LOCAL(dst + 8), 0, 0);
  __builtin_amdgcn_global_load_async_to_lds_b128(AS_GLOBAL(src + 16), AS_LOCAL(dst + 16), 0, 0);
  __builtin_amdgcn_global_load_async_to_lds_b128(AS_GLOBAL(src + 24), AS_LOCAL(dst + 24), 0, 0);
  (void)st;
#else
  const uint4* g = (const uint4*)src;
  st[0] = g[0]; st[1] = g[1]; st[2] = g[2]; st[3] = g[3];
  (void)dst;
#endif
}
__device__ __forceinline__ void cp64_commit(bf16_t* dst, const uint4* st) {
#if !GSER_ASYNC_COPY
  uint4* l = (uint4*)dst;
  l[0] = st[0]; l[1] = st[1]; l[2] = st[2]; l[3] = st[3];
#else
  (void)dst; (void)st;
#endif
}
__device__ __forceinline__ void cp_wait_all() {
#if GSER_ASYNC_COPY
  __builtin_amdgcn_s_wait_asynccnt(0);
#endif
}

// ---- A fragment: 16x32 bf16, M x K. lane: m = L&15, half = L>>4.
// elements 0..7  <- K = half*8 + [0,8); elements 8..15 <- K = 16 + half*8 + [0,8)
__device__ __forceinline__ v16bf load_a_frag(const bf16_t* rowbase, int lane) {
  int m = lane & 15, half = lane >> 4;
  const uint4* p = (const uint4*)(rowbase + m * LDSROW);
  union { uint4 u[2]; v16bf v; } t;
  t.u[0] = p[half];
  t.u[1] = p[2 + half];
  return t.v;
}

// ---- B fragment: 32x16 bf16, K x N. lane: n = L&15, half = L>>4.
// elements 0..7  <- K = half*8 + [0,8); elements 8..15 <- K = 16 + half*8 + [0,8)
__device__ __forceinline__ v16bf load_b_frag(const bf16_t* wsbase, int nh, int lane) {
  int n = lane & 15, half = lane >> 4;
  const uint4* p = (const uint4*)(wsbase + (nh * 16 + n) * LDSROW);
  union { uint4 u[2]; v16bf v; } t;
  t.u[0] = p[half];
  t.u[1] = p[2 + half];
  return t.v;
}

// ---------------------------------------------------------------------------
// f32 -> bf16 2D strided convert (8 elems / thread, b128 in / b128 out)
// ---------------------------------------------------------------------------
__global__ void gser_cvt2d(const float* __restrict__ src, bf16_t* __restrict__ dst,
                           int rows, int src_stride, int cols) {
  long id = (long)blockIdx.x * blockDim.x + threadIdx.x;
  int c8 = cols >> 3;
  long total = (long)rows * c8;
  if (id >= total) return;
  int r = (int)(id / c8);
  int c = (int)(id % c8) << 3;
  const float4* s = (const float4*)(src + (size_t)r * src_stride + c);
  float4 f0 = s[0], f1 = s[1];
  union { bf16_t b[8]; uint4 u; } t;
  t.b[0] = (bf16_t)f0.x; t.b[1] = (bf16_t)f0.y; t.b[2] = (bf16_t)f0.z; t.b[3] = (bf16_t)f0.w;
  t.b[4] = (bf16_t)f1.x; t.b[5] = (bf16_t)f1.y; t.b[6] = (bf16_t)f1.z; t.b[7] = (bf16_t)f1.w;
  *(uint4*)(dst + (size_t)r * cols + c) = t.u;
}

// ---------------------------------------------------------------------------
// W_res[:A,:A] -> bf16 transpose (dst[n][k] = src[k][n]) so every GEMM's
// B-operand is "row n = contiguous K".
// ---------------------------------------------------------------------------
__global__ void gser_transpose_cvt(const float* __restrict__ src, bf16_t* __restrict__ dst) {
  __shared__ float tile[32][33];
  int n = blockIdx.x * 32 + threadIdx.x;
  int k0 = blockIdx.y * 32;
#pragma unroll
  for (int j = 0; j < 32; j += 8)
    tile[threadIdx.y + j][threadIdx.x] = src[(size_t)(k0 + threadIdx.y + j) * MAXD + n];
  __syncthreads();
#pragma unroll
  for (int j = 0; j < 32; j += 8) {
    int dn = blockIdx.x * 32 + threadIdx.y + j;
    int dk = k0 + threadIdx.x;
    dst[(size_t)dn * AD + dk] = (bf16_t)tile[threadIdx.x][threadIdx.y + j];
  }
}

// ---------------------------------------------------------------------------
// Zero the padded half of the output: out[:, A:MAXD] = 0
// ---------------------------------------------------------------------------
__global__ void gser_zero_pad(float* __restrict__ out) {
  long id = (long)blockIdx.x * blockDim.x + threadIdx.x;
  const int c4 = (MAXD - AD) / 4;
  long total = (long)BDIM * c4;
  if (id >= total) return;
  int r = (int)(id / c4);
  int c = (int)(id % c4) * 4;
  *(float4*)(out + (size_t)r * MAXD + AD + c) = make_float4(0.f, 0.f, 0.f, 0.f);
}

// ---------------------------------------------------------------------------
// Fused main kernel. Grid: (BDIM/BM = 32, AD/BN = 32). Block: 256 (8 waves).
// Wave w owns rows [wgM*128 + w*16, +16) x cols [wgN*64, +64).
// Pipelined, double-buffered, one barrier per K-step:
//   phase 1 (16 steps): z += X@Win.T ; i/f/o += X@Wg.T      (16 WMMA / step)
//   phase 2 (64 steps): z += P@WresT.T                      ( 4 WMMA / step)
// Peeled boundary iteration keeps copies in flight across the phase switch
// (NT1 even -> LDS buffer parity is continuous).
// ---------------------------------------------------------------------------
__global__ void __launch_bounds__(256, 1)
gser_main(const bf16_t* __restrict__ Xb,     // inputs bf16   [BDIM][INDIM]
          const bf16_t* __restrict__ Pb,     // prev   bf16   [BDIM][AD]
          const bf16_t* __restrict__ Winb,   // W_in   bf16   [AD][INDIM]
          const bf16_t* __restrict__ Wgb,    // W_gate bf16   [3*AD][INDIM]
          const bf16_t* __restrict__ WresT,  // W_res^T bf16  [AD][AD]
          const float*  __restrict__ state,  // f32 [BDIM][MAXD] (exact prev)
          const float*  __restrict__ leak_param,
          const float*  __restrict__ thresh_param,
          float* __restrict__ out) {         // f32 [BDIM][MAXD]
  __shared__ alignas(16) bf16_t Xs[2][BM * LDSROW];       // 2 x 10240 B
  __shared__ alignas(16) bf16_t Ws[2][4][BN * LDSROW];    // 2 x 20480 B

  const int tid  = threadIdx.x;
  const int wid  = tid >> 5;
  const int lane = tid & 31;
  const int m0 = blockIdx.x * BM;
  const int n0 = blockIdx.y * BN;

  v8f accz[4] = {}; v8f acci[4] = {}; v8f accf[4] = {}; v8f acco[4] = {};

  // copy roles: X tile = 256 threads x 32B; weight tiles = 1 full 64B row each
  const int xr = tid >> 1, xh = tid & 1;    // 128 rows x 2 col-halves
  const int wgrp = tid >> 6;                // weight tile 0..3
  const int wrow = tid & 63;                // row within 64-row weight tile

  const bf16_t* xsrc1 = Xb + (size_t)(m0 + xr) * INDIM + xh * 16;
  const bf16_t* wsrc1 = (wgrp == 0)
      ? (Winb + (size_t)(n0 + wrow) * INDIM)
      : (Wgb + ((size_t)(wgrp - 1) * AD + n0 + wrow) * INDIM);
  const bf16_t* xsrc2 = Pb + (size_t)(m0 + xr) * AD + xh * 16;
  const bf16_t* wsrc2 = WresT + (size_t)(n0 + wrow) * AD;

  bf16_t* xdst[2] = { &Xs[0][xr * LDSROW + xh * 16], &Xs[1][xr * LDSROW + xh * 16] };
  bf16_t* wdst[2] = { &Ws[0][wgrp][wrow * LDSROW], &Ws[1][wgrp][wrow * LDSROW] };

  auto issue1 = [&](int t, uint4* stx, uint4* stw) {
    int b = t & 1, k0 = t * KT;
    cp32_issue(xdst[b], xsrc1 + k0, stx);
    cp64_issue(wdst[b], wsrc1 + k0, stw);
  };
  auto commit1 = [&](int t, uint4* stx, uint4* stw) {
    int b = t & 1;
    cp32_commit(xdst[b], stx);
    cp64_commit(wdst[b], stw);
  };
  auto issue2 = [&](int t, uint4* stx, uint4* stw) {
    int b = t & 1, k0 = t * KT;
    cp32_issue(xdst[b], xsrc2 + k0, stx);
    if (wgrp == 0) cp64_issue(wdst[b], wsrc2 + k0, stw);
  };
  auto commit2 = [&](int t, uint4* stx, uint4* stw) {
    int b = t & 1;
    cp32_commit(xdst[b], stx);
    if (wgrp == 0) cp64_commit(wdst[b], stw);
  };

  auto compute1 = [&](int b) {
    v16bf a = load_a_frag(&Xs[b][0] + (wid * 16) * LDSROW, lane);
#pragma unroll
    for (int nh = 0; nh < 4; ++nh) {
      accz[nh] = WMMA_BF16(a, load_b_frag(&Ws[b][0][0], nh, lane), accz[nh]);
      acci[nh] = WMMA_BF16(a, load_b_frag(&Ws[b][1][0], nh, lane), acci[nh]);
      accf[nh] = WMMA_BF16(a, load_b_frag(&Ws[b][2][0], nh, lane), accf[nh]);
      acco[nh] = WMMA_BF16(a, load_b_frag(&Ws[b][3][0], nh, lane), acco[nh]);
    }
  };
  auto compute2 = [&](int b) {
    v16bf a = load_a_frag(&Xs[b][0] + (wid * 16) * LDSROW, lane);
#pragma unroll
    for (int nh = 0; nh < 4; ++nh)
      accz[nh] = WMMA_BF16(a, load_b_frag(&Ws[b][0][0], nh, lane), accz[nh]);
  };

  // prologue: fill buffer 0
  {
    uint4 stx[2], stw[4];
    issue1(0, stx, stw);
    commit1(0, stx, stw);
    cp_wait_all();
    __syncthreads();
  }
  // phase 1 steady state
  for (int t = 0; t + 1 < NT1; ++t) {
    uint4 stx[2], stw[4];
    issue1(t + 1, stx, stw);
    compute1(t & 1);
    commit1(t + 1, stx, stw);
    cp_wait_all();
    __syncthreads();
  }
  // boundary: compute last phase-1 tile while phase-2 tile 0 streams in
  {
    uint4 stx[2], stw[4];
    issue2(0, stx, stw);
    compute1((NT1 - 1) & 1);
    commit2(0, stx, stw);
    cp_wait_all();
    __syncthreads();
  }
  // phase 2 steady state
  for (int t = 0; t + 1 < NT2; ++t) {
    uint4 stx[2], stw[4];
    issue2(t + 1, stx, stw);
    compute2(t & 1);
    commit2(t + 1, stx, stw);
    cp_wait_all();
    __syncthreads();
  }
  compute2((NT2 - 1) & 1);   // drain

  // ------------- Epilogue: gates + leaky tanh reservoir + soft threshold ----
  // C/D layout: lane n = L&15; VGPR r holds M = (L>>4)*8 + r.
  const int nl = lane & 15, rsel = lane >> 4;
#pragma unroll
  for (int nh = 0; nh < 4; ++nh) {
    int n = n0 + nh * 16 + nl;
    float leak   = sigmoidf_(leak_param[n]);
    float thresh = softplusf_(thresh_param[n]);
#pragma unroll
    for (int r = 0; r < 8; ++r) {
      int m = m0 + wid * 16 + rsel * 8 + r;
      float prev = state[(size_t)m * MAXD + n];
      float ig = sigmoidf_(acci[nh][r]);
      float fg = sigmoidf_(accf[nh][r]);
      float og = sigmoidf_(acco[nh][r]);
      float s = (1.0f - leak) * (fg * prev) + leak * tanhf(ig * accz[nh][r]);
      s *= og;
      s = (s > thresh) ? (s - thresh) : s;
      out[(size_t)m * MAXD + n] = s;
    }
  }
}

// ---------------------------------------------------------------------------
extern "C" void kernel_launch(void* const* d_in, const int* in_sizes, int n_in,
                              void* d_out, int out_size, void* d_ws, size_t ws_size,
                              hipStream_t stream) {
  (void)in_sizes; (void)n_in; (void)out_size; (void)ws_size;
  const float* inputs       = (const float*)d_in[0];  // [4096][512]
  const float* state        = (const float*)d_in[1];  // [4096][4096]
  const float* W_res        = (const float*)d_in[2];  // [4096][4096]
  const float* W_in         = (const float*)d_in[3];  // [4096][512]
  const float* W_gate       = (const float*)d_in[4];  // [12288][512]
  const float* leak_param   = (const float*)d_in[5];  // [4096]
  const float* thresh_param = (const float*)d_in[6];  // [4096]
  float* out = (float*)d_out;

  // bf16 workspace layout (36 MB total)
  char* ws = (char*)d_ws;
  size_t off = 0;
  bf16_t* Xb    = (bf16_t*)(ws + off); off += (size_t)BDIM * INDIM * sizeof(bf16_t);
  bf16_t* Pb    = (bf16_t*)(ws + off); off += (size_t)BDIM * AD    * sizeof(bf16_t);
  bf16_t* Winb  = (bf16_t*)(ws + off); off += (size_t)AD   * INDIM * sizeof(bf16_t);
  bf16_t* Wgb   = (bf16_t*)(ws + off); off += (size_t)3 * AD * INDIM * sizeof(bf16_t);
  bf16_t* WresT = (bf16_t*)(ws + off); off += (size_t)AD * AD * sizeof(bf16_t);

  auto blocks8 = [](long rows, long cols) {
    long t = rows * (cols >> 3);
    return (int)((t + 255) / 256);
  };

  gser_cvt2d<<<blocks8(BDIM, INDIM), 256, 0, stream>>>(inputs, Xb, BDIM, INDIM, INDIM);
  gser_cvt2d<<<blocks8(BDIM, AD),    256, 0, stream>>>(state,  Pb, BDIM, MAXD, AD);
  gser_cvt2d<<<blocks8(AD, INDIM),   256, 0, stream>>>(W_in,   Winb, AD, INDIM, INDIM);
  gser_cvt2d<<<blocks8(3 * AD, INDIM), 256, 0, stream>>>(W_gate, Wgb, 3 * AD, INDIM, INDIM);
  gser_transpose_cvt<<<dim3(AD / 32, AD / 32), dim3(32, 8), 0, stream>>>(W_res, WresT);

  gser_zero_pad<<<(int)(((long)BDIM * ((MAXD - AD) / 4) + 255) / 256), 256, 0, stream>>>(out);

  gser_main<<<dim3(BDIM / BM, AD / BN), 256, 0, stream>>>(
      Xb, Pb, Winb, Wgb, WresT, state, leak_param, thresh_param, out);
}
